// SharedSubnetwork_20564303413751
// MI455X (gfx1250) — matrix operand order repaired and checked by reading, not compiled, not run on hardware
//
#include <hip/hip_runtime.h>
#include <hip/hip_bf16.h>
#include <math.h>

// ---------------------------------------------------------------- constants
#define DIN   512      // D
#define HID   2048     // H
#define H3    6144     // 3H
#define SEQ   8192     // S
#define N1    1024     // H/2
#define N2    512      // H/4
#define N3    512      // D

#define BK    128      // K-chunk staged in LDS by the TDM

#define SCAN_BLOCKS  64
#define SCAN_THREADS 256

typedef __bf16 bf16;
typedef __attribute__((ext_vector_type(8)))  __bf16        v8bf;
typedef __attribute__((ext_vector_type(16))) __bf16        v16bf;
typedef __attribute__((ext_vector_type(8)))  float         v8f;
typedef __attribute__((ext_vector_type(4)))  unsigned int  v4u;
typedef __attribute__((ext_vector_type(8)))  int           v8i;
typedef __attribute__((ext_vector_type(4)))  int           v4i;

// ------------------------------------------------------- fragment loaders
// A-matrix 16x32 bf16 (ISA 7.12.2): lane l -> row m=l&15;
//   elems 0..7  : K = 8*(l>>4) + 0..7 ; elems 8..15 : K = 16 + 8*(l>>4) + 0..7
__device__ inline v16bf load_a_frag(const bf16* __restrict__ a, int ld, int lane) {
    const int m  = lane & 15;
    const int kh = (lane >> 4) << 3;            // 0 or 8
    const bf16* p = a + m * ld + kh;
    v8bf lo = *(const v8bf*)(p);
    v8bf hi = *(const v8bf*)(p + 16);
    v16bf r;
#pragma unroll
    for (int i = 0; i < 8; ++i) { r[i] = lo[i]; r[i + 8] = hi[i]; }
    return r;
}

// B-matrix 32x16 bf16: lane l -> col n=l&15; elems 0..7 : K = 8*(l>>4) + 0..7 ; elems 8..15 : K = 16 + 8*(l>>4) + 0..7
// W stored [N,K] row-major (PyTorch layout) => column n of B is row n of W.
__device__ inline v16bf load_b_frag(const bf16* __restrict__ w, int ld, int lane) {
    const int n  = lane & 15;
    const int kh = (lane >> 4) << 3;            // 0 or 8
    const bf16* p = w + (size_t)n * ld + kh;
    v8bf lo = *(const v8bf*)(p);
    v8bf hi = *(const v8bf*)(p + 16);
    v16bf r;
#pragma unroll
    for (int i = 0; i < 8; ++i) { r[i] = lo[i]; r[i + 8] = hi[i]; }
    return r;
}

// ------------------------------------------------------- TDM: 2-D bf16 tile -> LDS
// D# per CDNA5 ISA: group0 = {count=1, lds_addr,
// global_addr[56:0], type=2}; group1 = {mask=0, data_size=1(2B),
// tensor_dim0/1, tile_dim0/1, tensor_dim0_stride}; trailing groups unused (2-D).
// This toolchain exposes the 6-arg builtin:
//   (uint32x4 g0, int32x8 g1, int32x4, int32x4, int32x8, i32 cpol)
__device__ inline void tdm_load_tile_bf16(const bf16* gptr, unsigned lds_off,
                                          int tensor_w, int tensor_h,
                                          int tile_w, int tile_h, int row_stride) {
    unsigned long long ga = (unsigned long long)(uintptr_t)gptr;
    v4u g0;
    g0[0] = 1u;                                           // count = 1 (valid D#)
    g0[1] = lds_off;                                      // lds_addr (bytes)
    g0[2] = (unsigned)(ga & 0xffffffffu);                 // global_addr[31:0]
    g0[3] = (unsigned)((ga >> 32) & 0x01ffffffu) | (2u << 30); // addr[56:32] | type=2
    v8i g1;
    g1[0] = (int)(1u << 16);                              // data_size=1 -> 2 bytes
    g1[1] = (int)(((unsigned)tensor_w & 0xffffu) << 16);  // tensor_dim0[15:0]
    g1[2] = (int)(((unsigned)tensor_w >> 16) |
                  (((unsigned)tensor_h & 0xffffu) << 16));// dim0[31:16] | dim1[15:0]
    g1[3] = (int)(((unsigned)tensor_h >> 16) |
                  ((unsigned)tile_w << 16));              // dim1[31:16] | tile_dim0
    g1[4] = (int)((unsigned)tile_h);                      // tile_dim1 | tile_dim2=0
    g1[5] = (int)((unsigned)row_stride);                  // tensor_dim0_stride[31:0]
    g1[6] = 0;                                            // stride0[47:32] | stride1 lo
    g1[7] = 0;
    v4i g2 = {0, 0, 0, 0};
    v4i g3 = {0, 0, 0, 0};
    v8i g4 = {0, 0, 0, 0, 0, 0, 0, 0};
    __builtin_amdgcn_tensor_load_to_lds(g0, g1, g2, g3, g4, 0);
}

// ------------------------------------------------------- WMMA GEMM
// out[M,N] = A[M,K](bf16) @ W[N,K]^T(bf16) + bias
// mode 0: f32 output; mode 1: relu + bf16 output.
// grid = (N/512, M/16), block = 256 (8 waves, each: 4 N-tiles).
// Wave 0 TDM-stages the shared 16xBK A chunk into LDS (double buffered);
// all waves read A fragments from LDS and stream B columns from global/L2.
__global__ __launch_bounds__(256)
void gemm_wmma_bf16(const bf16* __restrict__ A, const bf16* __restrict__ W,
                    const float* __restrict__ bias,
                    float* __restrict__ outF, bf16* __restrict__ outH,
                    int M, int N, int K, int mode) {
    __shared__ __attribute__((aligned(128))) bf16 asmem[2][16 * BK];

    const int lane = threadIdx.x & 31;
    const int wave = threadIdx.x >> 5;
    const int row0 = blockIdx.y << 4;
    const int col0 = ((blockIdx.x << 3) + wave) << 6;   // 4 tiles * 16

    v8f acc[4];
#pragma unroll
    for (int j = 0; j < 4; ++j)
#pragma unroll
        for (int v = 0; v < 8; ++v) acc[j][v] = 0.0f;

    const bf16* aRow = A + (size_t)row0 * K;
    const unsigned lds0 = (unsigned)(uintptr_t)&asmem[0][0];
    const unsigned lds1 = (unsigned)(uintptr_t)&asmem[1][0];

    if (wave == 0)
        tdm_load_tile_bf16(aRow, lds0, K, M, BK, 16, K);

    int buf = 0;
    for (int kc = 0; kc < K; kc += BK) {
        if (wave == 0)
            __builtin_amdgcn_s_wait_tensorcnt(0);
        __syncthreads();               // buf ready AND everyone done with buf^1
        if (wave == 0 && kc + BK < K)
            tdm_load_tile_bf16(aRow + kc + BK, buf ? lds0 : lds1, K, M, BK, 16, K);

        const bf16* as = &asmem[buf][0];
#pragma unroll
        for (int ks = 0; ks < BK; ks += 32) {
            // batch all loads of this k-step into one clause, then 4 WMMAs
            v16bf bf4[4];
#pragma unroll
            for (int j = 0; j < 4; ++j)
                bf4[j] = load_b_frag(W + (size_t)(col0 + (j << 4)) * K + kc + ks,
                                     K, lane);
            v16bf af = load_a_frag(as + ks, BK, lane);
#pragma unroll
            for (int j = 0; j < 4; ++j)
                acc[j] = __builtin_amdgcn_wmma_f32_16x16x32_bf16(
                    false, af, false, bf4[j], (short)0, acc[j], false, false);
        }
        buf ^= 1;
    }

    // C/D layout: elem v -> M = row0 + v + 8*(lane>>4), N = col0 + j*16 + (lane&15)
    const int n  = lane & 15;
    const int hi = lane >> 4;
#pragma unroll
    for (int j = 0; j < 4; ++j) {
        const int col = col0 + (j << 4) + n;
        const float b = bias ? bias[col] : 0.0f;
#pragma unroll
        for (int v = 0; v < 8; ++v) {
            const int m = row0 + v + (hi << 3);
            float val = acc[j][v] + b;
            if (mode == 1) {
                val = val > 0.0f ? val : 0.0f;
                outH[(size_t)m * N + col] = (bf16)val;
            } else {
                outF[(size_t)m * N + col] = val;
            }
        }
    }
}

// ------------------------------------------------------- f32 -> bf16
__global__ void f32_to_bf16(const float* __restrict__ in, bf16* __restrict__ out, int n) {
    int i = blockIdx.x * blockDim.x + threadIdx.x;
    if (i < n) out[i] = (bf16)in[i];
}

// ------------------------------------------------------- scan init
__global__ void scan_init(float* h0, float* h1, int* bar) {
    int i = blockIdx.x * blockDim.x + threadIdx.x;
    if (i < HID) { h0[i] = 0.0f; h1[i] = 0.0f; }
    if (i == 0)  { bar[0] = 0; bar[1] = 0; }
}

// ------------------------------------------------------- persistent GRU scan
// 2048 groups x 8 lanes. Group g computes gh[g], gh[g+H], gh[g+2H] itself
// (so the gate update needs no intra-step barrier), updates h[g], then one
// device-wide barrier per step. W_hh is bf16 (25 MB -> L2-resident).
__global__ __launch_bounds__(SCAN_THREADS)
void gru_scan(const bf16* __restrict__ Whh, const float* __restrict__ bhh,
              const float* __restrict__ xproj,
              float* __restrict__ hbuf0, float* __restrict__ hbuf1,
              float* __restrict__ hs, bf16* __restrict__ hs16,
              int* __restrict__ bar_count, int* __restrict__ bar_gen) {
    const int tid = blockIdx.x * SCAN_THREADS + threadIdx.x;  // 0..16383
    const int g   = tid >> 3;                                 // hidden idx
    const int sub = tid & 7;
    const int k0  = sub * (HID / 8);                          // 256-wide slice

    const bf16* wr = Whh + (size_t)g * HID + k0;
    const bf16* wz = Whh + (size_t)(g + HID) * HID + k0;
    const bf16* wn = Whh + (size_t)(g + 2 * HID) * HID + k0;
    const float br = bhh[g], bz = bhh[g + HID], bn = bhh[g + 2 * HID];

    for (int t = 0; t < SEQ; ++t) {
        const float* hp = (t & 1) ? hbuf1 : hbuf0;
        float*       hn = (t & 1) ? hbuf0 : hbuf1;

        float dr = 0.0f, dz = 0.0f, dn = 0.0f;
        const float* hk = hp + k0;
#pragma unroll 8
        for (int k = 0; k < HID / 8; ++k) {
            const float hv = hk[k];
            dr += (float)wr[k] * hv;
            dz += (float)wz[k] * hv;
            dn += (float)wn[k] * hv;
        }
#pragma unroll
        for (int off = 4; off >= 1; off >>= 1) {
            dr += __shfl_xor(dr, off, 32);
            dz += __shfl_xor(dz, off, 32);
            dn += __shfl_xor(dn, off, 32);
        }
        if (sub == 0) {
            const float* xp = xproj + (size_t)t * H3;
            const float r  = 1.0f / (1.0f + __expf(-(xp[g] + dr + br)));
            const float z  = 1.0f / (1.0f + __expf(-(xp[g + HID] + dz + bz)));
            const float nn = tanhf(xp[g + 2 * HID] + r * (dn + bn));
            const float hv = (1.0f - z) * nn + z * hp[g];
            hn[g] = hv;
            hs[(size_t)t * HID + g]   = hv;
            hs16[(size_t)t * HID + g] = (bf16)hv;
        }
        // ---- device-wide barrier (monotone count + generation flag) ----
        __threadfence();
        __syncthreads();
        if (threadIdx.x == 0) {
            const int arrived = atomicAdd(bar_count, 1) + 1;
            if (arrived == SCAN_BLOCKS * (t + 1)) {
                __threadfence();
                __hip_atomic_store(bar_gen, t + 1, __ATOMIC_RELEASE,
                                   __HIP_MEMORY_SCOPE_AGENT);
            }
            while (__hip_atomic_load(bar_gen, __ATOMIC_ACQUIRE,
                                     __HIP_MEMORY_SCOPE_AGENT) <= t) {
                __builtin_amdgcn_s_sleep(2);
            }
        }
        __syncthreads();
    }
}

// ------------------------------------------------------- attention
// e[s] = hs[s,:] . attn_w[D:D+H]  (the last_x . attn_w[:D] + attn_b term is
// constant across s and cancels exactly in the softmax).
__global__ void attn_scores(const float* __restrict__ hs,
                            const float* __restrict__ attn_w,
                            float* __restrict__ e) {
    const int lane = threadIdx.x & 31;
    const int row  = blockIdx.x * (blockDim.x >> 5) + (threadIdx.x >> 5);
    if (row >= SEQ) return;
    const float* hr = hs + (size_t)row * HID;
    const float* w  = attn_w + DIN;
    float acc = 0.0f;
    for (int k = lane; k < HID; k += 32) acc += hr[k] * w[k];
#pragma unroll
    for (int off = 16; off >= 1; off >>= 1) acc += __shfl_xor(acc, off, 32);
    if (lane == 0) e[row] = acc;
}

__global__ __launch_bounds__(1024)
void attn_softmax(const float* __restrict__ e, float* __restrict__ p,
                  float* __restrict__ sum_out) {
    __shared__ float red[1024];
    const int tid = threadIdx.x;
    float m = -3.402823466e38f;
    for (int s = tid; s < SEQ; s += 1024) m = fmaxf(m, e[s]);
    red[tid] = m; __syncthreads();
    for (int off = 512; off >= 1; off >>= 1) {
        if (tid < off) red[tid] = fmaxf(red[tid], red[tid + off]);
        __syncthreads();
    }
    const float mx = red[0]; __syncthreads();
    float sm = 0.0f;
    for (int s = tid; s < SEQ; s += 1024) {
        const float v = __expf(e[s] - mx);
        p[s] = v; sm += v;
    }
    red[tid] = sm; __syncthreads();
    for (int off = 512; off >= 1; off >>= 1) {
        if (tid < off) red[tid] += red[tid + off];
        __syncthreads();
    }
    if (tid == 0) sum_out[0] = red[0];
}

__global__ void attn_context(const float* __restrict__ hs, const float* __restrict__ p,
                             const float* __restrict__ sum_out, float* __restrict__ ctx) {
    const int h = blockIdx.x * blockDim.x + threadIdx.x;
    if (h >= HID) return;
    const float inv = 1.0f / sum_out[0];
    float acc = 0.0f;
    for (int s = 0; s < SEQ; ++s) acc += p[s] * hs[(size_t)s * HID + h];
    ctx[h] = acc * inv;
}

// ------------------------------------------------------- host side
extern "C" void kernel_launch(void* const* d_in, const int* in_sizes, int n_in,
                              void* d_out, int out_size, void* d_ws, size_t ws_size,
                              hipStream_t stream) {
    const float* x     = (const float*)d_in[0];   // [1,S,D]
    const float* W_ih  = (const float*)d_in[2];   // [3H,D]
    const float* W_hh  = (const float*)d_in[3];   // [3H,H]
    const float* b_ih  = (const float*)d_in[4];
    const float* b_hh  = (const float*)d_in[5];
    const float* fc1w  = (const float*)d_in[6];   // [N1,H]
    const float* fc1b  = (const float*)d_in[7];
    const float* fc2w  = (const float*)d_in[8];   // [N2,N1]
    const float* fc2b  = (const float*)d_in[9];
    const float* fc3w  = (const float*)d_in[10];  // [N3,N2]
    const float* fc3b  = (const float*)d_in[11];
    const float* attnw = (const float*)d_in[12];  // [1,D+H]

    float* out = (float*)d_out;                   // [ctx(2048) | enc(S*D)]

    char* ws = (char*)d_ws;
    size_t off = 0;
    auto take = [&](size_t bytes) -> char* {
        char* p = ws + off;
        off = (off + bytes + 255) & ~(size_t)255;
        return p;
    };

    float* xproj = (float*)take((size_t)SEQ * H3 * 4);
    bf16*  x16   = (bf16*)take((size_t)SEQ * DIN * 2);
    bf16*  wih16 = (bf16*)take((size_t)H3 * DIN * 2);
    bf16*  whh16 = (bf16*)take((size_t)H3 * HID * 2);
    bf16*  f1w16 = (bf16*)take((size_t)N1 * HID * 2);
    bf16*  f2w16 = (bf16*)take((size_t)N2 * N1 * 2);
    bf16*  f3w16 = (bf16*)take((size_t)N3 * N2 * 2);
    float* hs    = (float*)take((size_t)SEQ * HID * 4);
    bf16*  hs16  = (bf16*)take((size_t)SEQ * HID * 2);
    bf16*  o1    = (bf16*)take((size_t)SEQ * N1 * 2);
    bf16*  o2    = (bf16*)take((size_t)SEQ * N2 * 2);
    float* h0    = (float*)take(HID * 4);
    float* h1    = (float*)take(HID * 4);
    float* e     = (float*)take(SEQ * 4);
    float* p     = (float*)take(SEQ * 4);
    float* ssum  = (float*)take(256);
    int*   bar   = (int*)take(256);

    auto cvt = [&](const float* src, bf16* dst, int n) {
        f32_to_bf16<<<(n + 255) / 256, 256, 0, stream>>>(src, dst, n);
    };

    // 0) init scan state
    scan_init<<<(HID + 255) / 256, 256, 0, stream>>>(h0, h1, bar);

    // 1) bf16 copies of activations and weights
    cvt(x,    x16,   SEQ * DIN);
    cvt(W_ih, wih16, H3 * DIN);
    cvt(W_hh, whh16, H3 * HID);
    cvt(fc1w, f1w16, N1 * HID);
    cvt(fc2w, f2w16, N2 * N1);
    cvt(fc3w, f3w16, N3 * N2);

    // 2) x_proj = x @ W_ih^T + b_ih   (WMMA + TDM-staged A)
    gemm_wmma_bf16<<<dim3(H3 / 512, SEQ / 16), 256, 0, stream>>>(
        x16, wih16, b_ih, xproj, nullptr, SEQ, H3, DIN, 0);

    // 3) sequential GRU scan (persistent, one grid barrier per step);
    //    emits hs in both f32 (attention) and bf16 (MLP)
    gru_scan<<<SCAN_BLOCKS, SCAN_THREADS, 0, stream>>>(
        whh16, b_hh, xproj, h0, h1, hs, hs16, bar, bar + 1);

    // 4) encoder MLP (WMMA, relu fused, bf16 between layers)
    gemm_wmma_bf16<<<dim3(N1 / 512, SEQ / 16), 256, 0, stream>>>(
        hs16, f1w16, fc1b, nullptr, o1, SEQ, N1, HID, 1);
    gemm_wmma_bf16<<<dim3(N2 / 512, SEQ / 16), 256, 0, stream>>>(
        o1, f2w16, fc2b, nullptr, o2, SEQ, N2, N1, 1);
    gemm_wmma_bf16<<<dim3(N3 / 512, SEQ / 16), 256, 0, stream>>>(
        o2, f3w16, fc3b, out + HID, nullptr, SEQ, N3, N2, 0);

    // 5) attention decoder -> context vector (first 2048 outputs)
    attn_scores<<<SEQ / 8, 256, 0, stream>>>(hs, attnw, e);
    attn_softmax<<<1, 1024, 0, stream>>>(e, p, ssum);
    attn_context<<<HID / 256, 256, 0, stream>>>(hs, p, ssum, out);
}
